// TaylorWindowAttention_7653631721884
// MI455X (gfx1250) — matrix operand order repaired and checked by reading, not compiled, not run on hardware
//
#include <hip/hip_runtime.h>
#include <hip/hip_bf16.h>
#include <stdint.h>

// ---------------- CDNA5 / gfx1250 Taylor-window attention ----------------
// GEMMs via v_wmma_f32_16x16x32_bf16 (fp32 accumulate).
// Block-level LDS staging of shared operands via async global->LDS copies
// (ASYNCcnt), double buffered across split barriers; each wave computes a
// 32x64 accumulator tile. Fragment loads are hoisted ahead of the WMMA
// chains so LDS latency overlaps matrix-pipe execution.

typedef __attribute__((ext_vector_type(16))) __bf16 v16bf;
typedef __attribute__((ext_vector_type(8)))  __bf16 v8bf;
typedef __attribute__((ext_vector_type(8)))  float  v8f;

#define DIMC   768
#define NHEADS 12
#define HDIM   64
#define BATCH  4
#define SEQ    1024
#define ROWS   (BATCH*SEQ)     /* 4096 */
#define QKVC   (3*DIMC)        /* 2304 */
#define QSCALE 0.125f          /* 64^-0.5 */

__device__ __forceinline__ __bf16 f2bf(float f) {
  unsigned u = __builtin_bit_cast(unsigned, f);
  u += 0x7FFFu + ((u >> 16) & 1u);            // round-to-nearest-even
  unsigned short s = (unsigned short)(u >> 16);
  return __builtin_bit_cast(__bf16, s);
}

// Build a 16-element bf16 fragment from two contiguous 8-element (16B) chunks.
__device__ __forceinline__ v16bf frag16(const __bf16* p0, const __bf16* p1) {
  v8bf a = *(const v8bf*)p0;
  v8bf b = *(const v8bf*)p1;
  return __builtin_shufflevector(a, b, 0,1,2,3,4,5,6,7,8,9,10,11,12,13,14,15);
}

__device__ __forceinline__ v8f wmma_bf16(v16bf a, v16bf b, v8f c) {
  return __builtin_amdgcn_wmma_f32_16x16x32_bf16(false, a, false, b, (short)0, c,
                                                 false, false);
}

// ---- async global -> LDS staging (CDNA5 ASYNCcnt path), 16B per lane ----
__device__ __forceinline__ void stage_chunk(const __bf16* g, __bf16* l) {
  unsigned loff = (unsigned)(size_t)l;                 // LDS byte address
  unsigned long long ga = (unsigned long long)(size_t)g;
  asm volatile("global_load_async_to_lds_b128 %0, %1, off"
               :: "v"(loff), "v"(ga) : "memory");
}
__device__ __forceinline__ void stage_fence() {        // my async stores landed
  asm volatile("s_wait_asynccnt 0x0" ::: "memory");
}

// ---------------- prep kernels ----------------

__global__ void k_cvt(const float* __restrict__ s, __bf16* __restrict__ d, int n) {
  int i = blockIdx.x * 256 + threadIdx.x;
  if (i < n) d[i] = f2bf(s[i]);
}

// s: [R][Cc] fp32 row-major  ->  d: [Cc][R] bf16 row-major (transposed)
__global__ void k_transpose(const float* __restrict__ s, __bf16* __restrict__ d,
                            int R, int Cc) {
  int i = blockIdx.x * 256 + threadIdx.x;
  if (i >= R * Cc) return;
  int j = i / R;
  int c = i % R;
  d[(size_t)j * R + c] = f2bf(s[(size_t)c * Cc + j]);
}

// stage a 64(col-of-W) x 32(k) bf16 tile of WT into LDS: 256 threads x 16B
__device__ __forceinline__ void stageB(__bf16* buf, const __bf16* WT,
                                       int j0, int k0) {
  const int t = threadIdx.x;
  const int row = t >> 2, ch = (t & 3) * 8;
  stage_chunk(WT + (size_t)(j0 + row) * DIMC + k0 + ch, buf + row * 32 + ch);
}

// ---------------- QKV GEMM: [4096,768] @ [768,2304] ----------------
// Block: 8 waves x 32 rows = 256 rows x 64 cols. B tile staged in LDS,
// double buffered. Tile width 64 == HDIM so a tile is one (q|k|v, head).
__global__ __launch_bounds__(256) void k_qkv(const __bf16* __restrict__ X,
                                             const __bf16* __restrict__ WT,
                                             const float*  __restrict__ bias,
                                             __bf16* __restrict__ Q,
                                             __bf16* __restrict__ Kmat,
                                             __bf16* __restrict__ VT) {
  __shared__ __bf16 bst[2][64 * 32];
  const int lane = threadIdx.x & 31, w = threadIdx.x >> 5;
  const int r = lane & 15, hi = lane >> 4;
  const int blockM = blockIdx.x % (ROWS / 256);        // 16
  const int tn     = blockIdx.x / (ROWS / 256);        // 36
  const int m0 = blockM * 256 + w * 32, j0 = tn * 64;

  v8f acc[2][4];
#pragma unroll
  for (int rt = 0; rt < 2; ++rt)
#pragma unroll
    for (int t = 0; t < 4; ++t)
#pragma unroll
      for (int g = 0; g < 8; ++g) acc[rt][t][g] = 0.f;

  const __bf16* arow0 = X + (size_t)(m0 + r) * DIMC;
  const __bf16* arow1 = X + (size_t)(m0 + 16 + r) * DIMC;

  stageB(bst[0], WT, j0, 0);
  int cur = 0;
  for (int k0 = 0; k0 < DIMC; k0 += 32) {
    stage_fence();
    __syncthreads();                                   // bst[cur] published
    if (k0 + 32 < DIMC) stageB(bst[cur ^ 1], WT, j0, k0 + 32);
    v16bf a0 = frag16(arow0 + k0 + hi * 8, arow0 + k0 + 16 + hi * 8);
    v16bf a1 = frag16(arow1 + k0 + hi * 8, arow1 + k0 + 16 + hi * 8);
    const __bf16* bt = bst[cur];
    // issue all four B-fragment LDS loads up front (one clause, one wait)
    v16bf bfm[4];
#pragma unroll
    for (int t = 0; t < 4; ++t) {
      const __bf16* brow = bt + (t * 16 + r) * 32 + hi * 8;
      bfm[t] = frag16(brow, brow + 16);
    }
#pragma unroll
    for (int t = 0; t < 4; ++t) {
      acc[0][t] = wmma_bf16(a0, bfm[t], acc[0][t]);
      acc[1][t] = wmma_bf16(a1, bfm[t], acc[1][t]);
    }
    cur ^= 1;
  }

  const int t3 = j0 / DIMC;                            // 0=q 1=k 2=v
  const int h  = (j0 % DIMC) / HDIM;
#pragma unroll
  for (int rt = 0; rt < 2; ++rt)
#pragma unroll
    for (int t = 0; t < 4; ++t) {
      const int d = t * 16 + r;
      const float bv = bias[j0 + t * 16 + r];
#pragma unroll
      for (int g = 0; g < 8; ++g) {
        const int row = m0 + rt * 16 + g + 8 * hi;
        const int b = row >> 10, n = row & 1023;
        const size_t bh = (size_t)b * NHEADS + h;
        const float val = acc[rt][t][g] + bv;
        if (t3 == 0)      Q   [(bh * SEQ + n) * HDIM + d] = f2bf(val * QSCALE);
        else if (t3 == 1) Kmat[(bh * SEQ + n) * HDIM + d] = f2bf(val);
        else              VT  [(bh * HDIM + d) * SEQ + n] = f2bf(val);
      }
    }
}

// stage K tile [32 m][64 d] + V^T tile [64 d][32 m]: 512 x 16B chunks
__device__ __forceinline__ void stageKV(__bf16* kbuf, __bf16* vbuf,
                                        const __bf16* kb, const __bf16* vb,
                                        int m0) {
  const int t = threadIdx.x;
  { const int row = t >> 3, ch = (t & 7) * 8;          // 32 rows x 8 chunks
    stage_chunk(kb + (size_t)(m0 + row) * HDIM + ch, kbuf + row * 64 + ch); }
  { const int row = t >> 2, ch = (t & 3) * 8;          // 64 rows x 4 chunks
    stage_chunk(vb + (size_t)row * SEQ + m0 + ch, vbuf + row * 32 + ch); }
}

// ---------------- fused Taylor attention ----------------
// Block = 8 waves x 32 q-rows = 256 rows of one head; K/V tiles staged in
// LDS (double buffered) and shared by all 8 waves. grid = 48 heads x 4.
__global__ __launch_bounds__(256) void k_attn(const __bf16* __restrict__ Q,
                                              const __bf16* __restrict__ Kmat,
                                              const __bf16* __restrict__ VT,
                                              __bf16* __restrict__ AO) {
  __shared__ __bf16 kvs[2][32 * 64 + 64 * 32];         // 2 x 8KB K+V tiles
  __shared__ __bf16 scr[8][32 * 32];                   // per-wave score relayout
  const int lane = threadIdx.x & 31, w = threadIdx.x >> 5;
  const int r = lane & 15, hi = lane >> 4;
  const int bh   = blockIdx.x >> 2;                    // 0..47
  const int nblk = blockIdx.x & 3;
  const int n0 = nblk * 256 + w * 32;

  const __bf16* qb = Q    + (size_t)bh * SEQ * HDIM;
  const __bf16* kb = Kmat + (size_t)bh * SEQ * HDIM;
  const __bf16* vb = VT   + (size_t)bh * HDIM * SEQ;

  // q fragments: 2 row-tiles x 2 k-chunks (K = d)
  v16bf aq[2][2];
#pragma unroll
  for (int rt = 0; rt < 2; ++rt) {
    const __bf16* qrow = qb + (size_t)(n0 + rt * 16 + r) * HDIM;
    aq[rt][0] = frag16(qrow + hi * 8,      qrow + 16 + hi * 8);
    aq[rt][1] = frag16(qrow + 32 + hi * 8, qrow + 48 + hi * 8);
  }

  v8f acc[2][4];
  float dsum[2][8];
#pragma unroll
  for (int rt = 0; rt < 2; ++rt) {
#pragma unroll
    for (int t = 0; t < 4; ++t)
#pragma unroll
      for (int g = 0; g < 8; ++g) acc[rt][t][g] = 0.f;
#pragma unroll
    for (int g = 0; g < 8; ++g) dsum[rt][g] = 0.f;
  }

  __bf16* myscr = scr[w];
  stageKV(kvs[0], kvs[0] + 32 * 64, kb, vb, 0);
  int cur = 0;

  for (int m0 = 0; m0 < SEQ; m0 += 32) {
    stage_fence();
    __syncthreads();                                   // kvs[cur] published
    const __bf16* kt = kvs[cur];
    const __bf16* vt = kvs[cur] + 32 * 64;
    if (m0 + 32 < SEQ)
      stageKV(kvs[cur ^ 1], kvs[cur ^ 1] + 32 * 64, kb, vb, m0 + 32);

    // hoist all K and V fragment loads: their LDS latency overlaps the
    // score WMMAs and the Taylor VALU work below
    v16bf bk[2][2];
#pragma unroll
    for (int sub = 0; sub < 2; ++sub) {
      const __bf16* krow = kt + (sub * 16 + r) * 64 + hi * 8;
      bk[sub][0] = frag16(krow, krow + 16);
      bk[sub][1] = frag16(krow + 32, krow + 48);
    }
    v16bf bv[4];
#pragma unroll
    for (int t = 0; t < 4; ++t) {
      const __bf16* vrow = vt + (t * 16 + r) * 32 + hi * 8;
      bv[t] = frag16(vrow, vrow + 16);
    }

    // ---- scores + Taylor numerator ----
#pragma unroll
    for (int sub = 0; sub < 2; ++sub) {
#pragma unroll
      for (int rt = 0; rt < 2; ++rt) {
        v8f s;
#pragma unroll
        for (int g = 0; g < 8; ++g) s[g] = 0.f;
        s = wmma_bf16(aq[rt][0], bk[sub][0], s);
        s = wmma_bf16(aq[rt][1], bk[sub][1], s);
#pragma unroll
        for (int g = 0; g < 8; ++g) {
          const float a = s[g];
          const float num = 1.0f + a + 0.5f * a * a;   // 0.5(a+1)^2+0.5 > 0
          dsum[rt][g] += num;
          myscr[(rt * 16 + g + 8 * hi) * 32 + sub * 16 + r] = f2bf(num);
        }
      }
    }
    // C-layout stores -> A-layout loads (wave private): split-counter fence
    asm volatile("s_wait_dscnt 0x0" ::: "memory");

    // ---- numerator @ V ----
#pragma unroll
    for (int rt = 0; rt < 2; ++rt) {
      const __bf16* arow = myscr + (rt * 16 + r) * 32;
      v16bf an = frag16(arow + hi * 8, arow + 16 + hi * 8);
#pragma unroll
      for (int t = 0; t < 4; ++t)
        acc[rt][t] = wmma_bf16(an, bv[t], acc[rt][t]);
    }
    cur ^= 1;
  }

  // row sums live across 16 lanes of each half-wave -> butterfly reduce
  const int b = bh / NHEADS, h = bh % NHEADS;
#pragma unroll
  for (int rt = 0; rt < 2; ++rt) {
    float inv[8];
#pragma unroll
    for (int g = 0; g < 8; ++g) {
      float s = dsum[rt][g];
#pragma unroll
      for (int off = 1; off < 16; off <<= 1) s += __shfl_xor(s, off, 16);
      inv[g] = 1.0f / s;
    }
#pragma unroll
    for (int t = 0; t < 4; ++t) {
      const int d = t * 16 + r;
#pragma unroll
      for (int g = 0; g < 8; ++g) {
        const int n = n0 + rt * 16 + g + 8 * hi;
        AO[((size_t)(b * SEQ + n)) * DIMC + h * HDIM + d] =
            f2bf(acc[rt][t][g] * inv[g]);
      }
    }
  }
}

// ---------------- output projection: [4096,768] @ [768,768] + bias ----------------
__global__ __launch_bounds__(256) void k_proj(const __bf16* __restrict__ A,
                                              const __bf16* __restrict__ WT,
                                              const float*  __restrict__ bias,
                                              float* __restrict__ out) {
  __shared__ __bf16 bst[2][64 * 32];
  const int lane = threadIdx.x & 31, w = threadIdx.x >> 5;
  const int r = lane & 15, hi = lane >> 4;
  const int blockM = blockIdx.x % (ROWS / 256);        // 16
  const int tn     = blockIdx.x / (ROWS / 256);        // 12
  const int m0 = blockM * 256 + w * 32, j0 = tn * 64;

  v8f acc[2][4];
#pragma unroll
  for (int rt = 0; rt < 2; ++rt)
#pragma unroll
    for (int t = 0; t < 4; ++t)
#pragma unroll
      for (int g = 0; g < 8; ++g) acc[rt][t][g] = 0.f;

  const __bf16* arow0 = A + (size_t)(m0 + r) * DIMC;
  const __bf16* arow1 = A + (size_t)(m0 + 16 + r) * DIMC;

  stageB(bst[0], WT, j0, 0);
  int cur = 0;
  for (int k0 = 0; k0 < DIMC; k0 += 32) {
    stage_fence();
    __syncthreads();
    if (k0 + 32 < DIMC) stageB(bst[cur ^ 1], WT, j0, k0 + 32);
    v16bf a0 = frag16(arow0 + k0 + hi * 8, arow0 + k0 + 16 + hi * 8);
    v16bf a1 = frag16(arow1 + k0 + hi * 8, arow1 + k0 + 16 + hi * 8);
    const __bf16* bt = bst[cur];
    v16bf bfm[4];
#pragma unroll
    for (int t = 0; t < 4; ++t) {
      const __bf16* brow = bt + (t * 16 + r) * 32 + hi * 8;
      bfm[t] = frag16(brow, brow + 16);
    }
#pragma unroll
    for (int t = 0; t < 4; ++t) {
      acc[0][t] = wmma_bf16(a0, bfm[t], acc[0][t]);
      acc[1][t] = wmma_bf16(a1, bfm[t], acc[1][t]);
    }
    cur ^= 1;
  }
#pragma unroll
  for (int rt = 0; rt < 2; ++rt)
#pragma unroll
    for (int t = 0; t < 4; ++t) {
      const int col = j0 + t * 16 + r;
      const float bv = bias[col];
#pragma unroll
      for (int g = 0; g < 8; ++g) {
        const int row = m0 + rt * 16 + g + 8 * hi;
        out[(size_t)row * DIMC + col] = acc[rt][t][g] + bv;
      }
    }
}

// ---------------- host launcher ----------------
extern "C" void kernel_launch(void* const* d_in, const int* in_sizes, int n_in,
                              void* d_out, int out_size, void* d_ws, size_t ws_size,
                              hipStream_t stream) {
  const float* x     = (const float*)d_in[0];
  const float* wqkv  = (const float*)d_in[1];
  const float* bqkv  = (const float*)d_in[2];
  const float* wproj = (const float*)d_in[3];
  const float* bproj = (const float*)d_in[4];
  float* out = (float*)d_out;

  char* ws = (char*)d_ws;
  size_t off = 0;
  auto carve = [&](size_t bytes) {
    char* p = ws + off;
    off += (bytes + 255) & ~(size_t)255;
    return p;
  };
  __bf16* xbf    = (__bf16*)carve((size_t)ROWS * DIMC * 2);
  __bf16* wqkvT  = (__bf16*)carve((size_t)QKVC * DIMC * 2);
  __bf16* wprojT = (__bf16*)carve((size_t)DIMC * DIMC * 2);
  __bf16* Q      = (__bf16*)carve((size_t)ROWS * DIMC * 2);
  __bf16* Kmat   = (__bf16*)carve((size_t)ROWS * DIMC * 2);
  __bf16* VT     = (__bf16*)carve((size_t)ROWS * DIMC * 2);
  __bf16* AO     = (__bf16*)carve((size_t)ROWS * DIMC * 2);
  (void)ws_size; (void)n_in; (void)in_sizes; (void)out_size;

  k_cvt<<<(ROWS * DIMC + 255) / 256, 256, 0, stream>>>(x, xbf, ROWS * DIMC);
  k_transpose<<<(DIMC * QKVC + 255) / 256, 256, 0, stream>>>(wqkv, wqkvT, DIMC, QKVC);
  k_transpose<<<(DIMC * DIMC + 255) / 256, 256, 0, stream>>>(wproj, wprojT, DIMC, DIMC);

  k_qkv <<<(ROWS / 256) * (QKVC / 64), 256, 0, stream>>>(xbf, wqkvT, bqkv, Q, Kmat, VT);
  k_attn<<<48 * 4,                     256, 0, stream>>>(Q, Kmat, VT, AO);
  k_proj<<<(ROWS / 256) * (DIMC / 64), 256, 0, stream>>>(AO, wprojT, bproj, out);
}
